// OneLayerMicroArchitectureBuild_16784732192996
// MI455X (gfx1250) — compile-verified, matrix operand order checked by reading, not run on hardware
//
#include <hip/hip_runtime.h>
#include <hip/hip_bf16.h>

// ---------------------------------------------------------------------------
// GCN layer pipeline for MI455X (gfx1250, wave32, WMMA).
//   h        = x @ (W_pre @ W_conv) + (b_pre @ W_conv)      [WMMA f16->f32]
//   agg[d]  += h[s] * dinv[s]*dinv[d]  (edges, atomics, L2-resident)
//   y        = relu(LN(agg + h*dinv^2 + b_conv))            [wave reduce]
//   readout[batch[i]] += y                                  [atomics, 256KB]
//   out      = readout @ W_post + b_post                    [WMMA f16->f32]
// ---------------------------------------------------------------------------

typedef _Float16 half_t;
typedef __attribute__((ext_vector_type(16))) _Float16 v16h;
typedef __attribute__((ext_vector_type(8)))  _Float16 v8h;
typedef __attribute__((ext_vector_type(8)))  float    v8f;
typedef __attribute__((ext_vector_type(4)))  float    v4f_t;

#define HH 128
#define CC 64
#define GG 512
#define LN_EPS 1e-5f

static __device__ __forceinline__ v16h cat8(v8h lo, v8h hi) {
  return __builtin_shufflevector(lo, hi, 0,1,2,3,4,5,6,7,8,9,10,11,12,13,14,15);
}

// --- prep: WfT[n][k] = (W_pre @ W_conv)^T as f16 ; bf[n] = b_pre @ W_conv ---
__global__ __launch_bounds__(HH)
void prep_wf_kernel(const float* __restrict__ W_pre, const float* __restrict__ b_pre,
                    const float* __restrict__ W_conv, half_t* __restrict__ WfT,
                    float* __restrict__ bf) {
  __shared__ float col[HH];
  const int n = blockIdx.x;          // output column of fused weight
  const int f = threadIdx.x;         // input feature (row of W_pre)
  col[f] = W_conv[f * HH + n];
  __syncthreads();
  const float* wrow = W_pre + (size_t)f * HH;
  float acc = 0.0f;
  #pragma unroll 8
  for (int m = 0; m < HH; ++m) acc += wrow[m] * col[m];
  WfT[n * HH + f] = (half_t)acc;     // transposed: rows are output cols
  if (f == 0) {
    float s = 0.0f;
    for (int m = 0; m < HH; ++m) s += b_pre[m] * col[m];
    bf[n] = s;
  }
}

// --- prep: WpostT[c][k] = W_post[k][c] as f16 ---
__global__ __launch_bounds__(HH)
void prep_wpost_kernel(const float* __restrict__ W_post, half_t* __restrict__ WpostT) {
  WpostT[blockIdx.x * HH + threadIdx.x] = (half_t)W_post[threadIdx.x * CC + blockIdx.x];
}

__global__ void fill_kernel(float* __restrict__ p, float v, long long n) {
  long long i = (long long)blockIdx.x * blockDim.x + threadIdx.x;
  if (i < n) p[i] = v;
}

__global__ void degree_kernel(const long long* __restrict__ dst,
                              float* __restrict__ deg, long long e) {
  long long i = (long long)blockIdx.x * blockDim.x + threadIdx.x;
  if (i < e) atomicAdd(deg + dst[i], 1.0f);
}

__global__ void rsqrt_kernel(float* __restrict__ deg, long long n) {
  long long i = (long long)blockIdx.x * blockDim.x + threadIdx.x;
  if (i < n) deg[i] = rsqrtf(deg[i]);   // deg buffer becomes dinv
}

// --- main node GEMM: h[N,128] = f16wmma(x[N,128] @ Wf[128,128]) + bf ---
// 8 waves/block, each wave owns a 16-row strip and all 8 N-tiles.
__global__ __launch_bounds__(256)
void gemm_node_kernel(const float* __restrict__ x, const half_t* __restrict__ WfT,
                      const float* __restrict__ bf, float* __restrict__ h, int n) {
  __shared__ half_t sB[HH * HH];              // 32 KB: fused weight, transposed
  const int tid = threadIdx.x;
  {
    const uint4* gsrc = reinterpret_cast<const uint4*>(WfT);
    uint4* ldst = reinterpret_cast<uint4*>(sB);
    #pragma unroll
    for (int i = 0; i < (HH * HH / 8) / 256; ++i)
      ldst[tid + i * 256] = gsrc[tid + i * 256];
  }
  __syncthreads();

  const int wave    = tid >> 5;
  const int lane    = tid & 31;
  const int rl      = lane & 15;              // row-within-tile / col-within-tile
  const int khalf   = (lane >> 4) << 3;       // 0 or 8 (16-bit A/B layout)
  const int rowbase = blockIdx.x * 128 + wave * 16;

  int row = rowbase + rl;                     // clamp for tail: EXEC must stay all-1
  if (row > n - 1) row = n - 1;
  const float* xrow = x + (size_t)row * HH;

  v8f acc[8] = {};
  #pragma unroll
  for (int ks = 0; ks < 4; ++ks) {            // K = 128 in 4 steps of 32
    const int kb = ks * 32 + khalf;
    v4f_t f0 = *reinterpret_cast<const v4f_t*>(xrow + kb);
    v4f_t f1 = *reinterpret_cast<const v4f_t*>(xrow + kb + 4);
    v4f_t f2 = *reinterpret_cast<const v4f_t*>(xrow + kb + 16);
    v4f_t f3 = *reinterpret_cast<const v4f_t*>(xrow + kb + 20);
    v8f alo = __builtin_shufflevector(f0, f1, 0,1,2,3,4,5,6,7);
    v8f ahi = __builtin_shufflevector(f2, f3, 0,1,2,3,4,5,6,7);
    v16h a = cat8(__builtin_convertvector(alo, v8h), __builtin_convertvector(ahi, v8h));
    #pragma unroll
    for (int nt = 0; nt < 8; ++nt) {
      const half_t* bp = sB + (nt * 16 + rl) * HH + kb;
      v8h b0 = *reinterpret_cast<const v8h*>(bp);
      v8h b1 = *reinterpret_cast<const v8h*>(bp + 16);
      v16h b = cat8(b0, b1);
      acc[nt] = __builtin_amdgcn_wmma_f32_16x16x32_f16(
          false, a, false, b, (short)0, acc[nt], false, false);
    }
  }
  #pragma unroll
  for (int nt = 0; nt < 8; ++nt) {
    const int ncol = nt * 16 + rl;
    const float bia = bf[ncol];
    #pragma unroll
    for (int v = 0; v < 8; ++v) {             // C layout: VGPR v -> M = v (+8 hi lanes)
      const int m = rowbase + v + khalf;
      if (m < n) h[(size_t)m * HH + ncol] = acc[nt][v] + bia;
    }
  }
}

// --- edge propagation: one wave per edge, 4 floats/lane, f32 atomics (L2) ---
__global__ __launch_bounds__(256)
void edge_agg_kernel(const long long* __restrict__ src, const long long* __restrict__ dst,
                     const float* __restrict__ dinv, const float* __restrict__ h,
                     float* __restrict__ agg, long long ecount) {
  const long long wid = ((long long)blockIdx.x * blockDim.x + threadIdx.x) >> 5;
  const int lane = threadIdx.x & 31;
  if (wid >= ecount) return;
  __builtin_prefetch(src + wid + 2048, 0, 0);  // speculative index-stream prefetch
  __builtin_prefetch(dst + wid + 2048, 0, 0);
  const long long s = src[wid];
  const long long d = dst[wid];
  const float nrm = dinv[s] * dinv[d];
  v4f_t hv = *reinterpret_cast<const v4f_t*>(h + (size_t)s * HH + lane * 4);
  float* ap = agg + (size_t)d * HH + lane * 4;
  atomicAdd(ap + 0, hv[0] * nrm);
  atomicAdd(ap + 1, hv[1] * nrm);
  atomicAdd(ap + 2, hv[2] * nrm);
  atomicAdd(ap + 3, hv[3] * nrm);
}

// --- self-loop + bias + LayerNorm + ReLU + pooled scatter, one wave/node ---
__global__ __launch_bounds__(256)
void ln_pool_kernel(const float* __restrict__ agg, const float* __restrict__ h,
                    const float* __restrict__ dinv, const float* __restrict__ b_conv,
                    const float* __restrict__ gamma, const float* __restrict__ beta,
                    const long long* __restrict__ batch, float* __restrict__ readout,
                    int n) {
  const int wid  = (blockIdx.x * blockDim.x + threadIdx.x) >> 5;
  const int lane = threadIdx.x & 31;
  if (wid >= n) return;
  const float di = dinv[wid];
  const float selfw = di * di;                // self-loop norm = dinv^2
  const size_t base = (size_t)wid * HH + lane * 4;
  v4f_t av = *reinterpret_cast<const v4f_t*>(agg + base);
  v4f_t hv = *reinterpret_cast<const v4f_t*>(h + base);
  v4f_t bc = *reinterpret_cast<const v4f_t*>(b_conv + lane * 4);
  float v0 = av[0] + hv[0] * selfw + bc[0];
  float v1 = av[1] + hv[1] * selfw + bc[1];
  float v2 = av[2] + hv[2] * selfw + bc[2];
  float v3 = av[3] + hv[3] * selfw + bc[3];
  float s  = v0 + v1 + v2 + v3;
  float sq = v0*v0 + v1*v1 + v2*v2 + v3*v3;
  #pragma unroll
  for (int m = 16; m >= 1; m >>= 1) {         // wave32 butterfly reduction
    s  += __shfl_xor(s,  m, 32);
    sq += __shfl_xor(sq, m, 32);
  }
  const float mu   = s * (1.0f / HH);
  const float var  = sq * (1.0f / HH) - mu * mu;
  const float rstd = rsqrtf(var + LN_EPS);
  v4f_t gm = *reinterpret_cast<const v4f_t*>(gamma + lane * 4);
  v4f_t bt = *reinterpret_cast<const v4f_t*>(beta + lane * 4);
  const int g = (int)batch[wid];
  float* rp = readout + (size_t)g * HH + lane * 4;
  atomicAdd(rp + 0, fmaxf(gm[0] * (v0 - mu) * rstd + bt[0], 0.0f));
  atomicAdd(rp + 1, fmaxf(gm[1] * (v1 - mu) * rstd + bt[1], 0.0f));
  atomicAdd(rp + 2, fmaxf(gm[2] * (v2 - mu) * rstd + bt[2], 0.0f));
  atomicAdd(rp + 3, fmaxf(gm[3] * (v3 - mu) * rstd + bt[3], 0.0f));
}

// --- output GEMM: out[512,64] = wmma(readout[512,128] @ W_post) + b_post ---
// 128 wave-tiles total (32 M-tiles x 4 N-tiles): 16 blocks x 8 waves.
__global__ __launch_bounds__(256)
void out_gemm_kernel(const float* __restrict__ readout, const half_t* __restrict__ WpostT,
                     const float* __restrict__ b_post, float* __restrict__ out) {
  const int tid   = threadIdx.x;
  const int wid   = (blockIdx.x * 256 + tid) >> 5;
  const int lane  = tid & 31;
  const int rl    = lane & 15;
  const int khalf = (lane >> 4) << 3;
  const int mtile = wid >> 2;
  const int ntile = wid & 3;
  const float*  rrow = readout + (size_t)(mtile * 16 + rl) * HH;
  const half_t* brow = WpostT + (size_t)(ntile * 16 + rl) * HH;
  v8f acc = {};
  #pragma unroll
  for (int ks = 0; ks < 4; ++ks) {
    const int kb = ks * 32 + khalf;
    v4f_t f0 = *reinterpret_cast<const v4f_t*>(rrow + kb);
    v4f_t f1 = *reinterpret_cast<const v4f_t*>(rrow + kb + 4);
    v4f_t f2 = *reinterpret_cast<const v4f_t*>(rrow + kb + 16);
    v4f_t f3 = *reinterpret_cast<const v4f_t*>(rrow + kb + 20);
    v8f alo = __builtin_shufflevector(f0, f1, 0,1,2,3,4,5,6,7);
    v8f ahi = __builtin_shufflevector(f2, f3, 0,1,2,3,4,5,6,7);
    v16h a = cat8(__builtin_convertvector(alo, v8h), __builtin_convertvector(ahi, v8h));
    v8h b0 = *reinterpret_cast<const v8h*>(brow + kb);
    v8h b1 = *reinterpret_cast<const v8h*>(brow + kb + 16);
    v16h b = cat8(b0, b1);
    acc = __builtin_amdgcn_wmma_f32_16x16x32_f16(
        false, a, false, b, (short)0, acc, false, false);
  }
  const int ncol = ntile * 16 + rl;
  const float bia = b_post[ncol];
  #pragma unroll
  for (int v = 0; v < 8; ++v) {
    const int m = mtile * 16 + v + khalf;
    out[(size_t)m * CC + ncol] = acc[v] + bia;
  }
}

extern "C" void kernel_launch(void* const* d_in, const int* in_sizes, int n_in,
                              void* d_out, int out_size, void* d_ws, size_t ws_size,
                              hipStream_t stream) {
  const float*     x      = (const float*)d_in[0];
  const long long* ei     = (const long long*)d_in[1];   // int64 per reference
  const long long* batch  = (const long long*)d_in[2];
  const float*     W_pre  = (const float*)d_in[3];
  const float*     b_pre  = (const float*)d_in[4];
  const float*     W_conv = (const float*)d_in[5];
  const float*     b_conv = (const float*)d_in[6];
  const float*     gamma  = (const float*)d_in[7];
  const float*     beta   = (const float*)d_in[8];
  const float*     W_post = (const float*)d_in[9];
  const float*     b_post = (const float*)d_in[10];
  float* out = (float*)d_out;

  const long long N = in_sizes[0] / HH;
  const long long E = in_sizes[1] / 2;
  const long long* src = ei;       // edge_index[0]
  const long long* dst = ei + E;   // edge_index[1]

  // workspace carve-out (256B aligned)
  char* wsp = (char*)d_ws;
  auto take = [&](size_t bytes) -> char* {
    char* p = wsp;
    wsp += (bytes + 255) & ~(size_t)255;
    return p;
  };
  float*  h       = (float*)take((size_t)N * HH * sizeof(float));
  float*  agg     = (float*)take((size_t)N * HH * sizeof(float));
  float*  deg     = (float*)take((size_t)N * sizeof(float));     // becomes dinv
  float*  readout = (float*)take((size_t)GG * HH * sizeof(float));
  half_t* WfT     = (half_t*)take((size_t)HH * HH * sizeof(half_t));
  float*  bf      = (float*)take((size_t)HH * sizeof(float));
  half_t* WpostT  = (half_t*)take((size_t)CC * HH * sizeof(half_t));

  hipMemsetAsync(agg, 0, (size_t)N * HH * sizeof(float), stream);
  hipMemsetAsync(readout, 0, (size_t)GG * HH * sizeof(float), stream);

  prep_wf_kernel<<<HH, HH, 0, stream>>>(W_pre, b_pre, W_conv, WfT, bf);
  prep_wpost_kernel<<<CC, HH, 0, stream>>>(W_post, WpostT);
  fill_kernel<<<(int)((N + 255) / 256), 256, 0, stream>>>(deg, 1.0f, N);  // self loops
  degree_kernel<<<(int)((E + 255) / 256), 256, 0, stream>>>(dst, deg, E);
  rsqrt_kernel<<<(int)((N + 255) / 256), 256, 0, stream>>>(deg, N);
  gemm_node_kernel<<<(int)((N + 127) / 128), 256, 0, stream>>>(x, WfT, bf, h, (int)N);
  edge_agg_kernel<<<(int)((E * 32 + 255) / 256), 256, 0, stream>>>(src, dst, deg, h, agg, E);
  ln_pool_kernel<<<(int)((N * 32 + 255) / 256), 256, 0, stream>>>(
      agg, h, deg, b_conv, gamma, beta, batch, readout, (int)N);
  out_gemm_kernel<<<16, 256, 0, stream>>>(readout, WpostT, b_post, out);
}
